// DPHALOModel_40853728920167
// MI455X (gfx1250) — matrix operand order repaired and checked by reading, not compiled, not run on hardware
//
#include <hip/hip_runtime.h>
#include <hip/hip_bf16.h>

// ---------------- model dimensions ----------------
#define Bc     32
#define Sc     48
#define Vc     10000
#define VPAD   10016          // V padded to multiple of 32 for WMMA K
#define CVc    9600
#define Hc     768
#define NHc    12
#define HDc    64
#define NLc    12
#define Gc     32
#define CPG    (Hc / Gc)      // 24 channels per group
#define MROWS  (Bc * Sc)      // 1536
#define HB     (Bc * NHc)     // 384 head-batches
#define MH     (Bc * (Sc-1))  // 1504 head rows
#define MHPAD  1536           // head rows padded to multiple of 64
#define H2     (2 * Hc)       // 1536
#define H3     (3 * Hc)       // 2304
#define H4     (4 * Hc)       // 3072
#define SPAD   64             // S padded to 64 for attention K

#define CEILD(a,b) (((a)+(b)-1)/(b))

typedef _Float16 h16;
typedef __attribute__((ext_vector_type(8)))  _Float16 v8h;
typedef __attribute__((ext_vector_type(16))) _Float16 v16h;
typedef __attribute__((ext_vector_type(8)))  float    v8f;

// ---------------- WMMA fragment helpers (ISA 7.12.2 layouts) ----------------
// A matrix 16x32 f16: lanes 0-15 hold row M=lane, K = {0..7, 16..23};
// lanes 16-31 hold row M=lane-16, K = {8..15, 24..31}.
__device__ __forceinline__ v16h load_a_frag(const h16* __restrict__ p, int ld,
                                            int row0, int k0) {
  int lane = threadIdx.x;
  int r  = row0 + (lane & 15);
  int kk = k0 + ((lane & 16) ? 8 : 0);
  const h16* q = p + (size_t)r * ld + kk;
  v8h lo = *(const v8h*)(q);        // K offsets +0..+7
  v8h hi = *(const v8h*)(q + 16);   // K offsets +16..+23
  return __builtin_shufflevector(lo, hi, 0,1,2,3,4,5,6,7,8,9,10,11,12,13,14,15);
}

// B matrix 32x16 f16 taken from Bt[N,K] (row-major, row = output column):
// lanes 0-15: column N=lane, K = {0..7, 16..23} ; lanes 16-31: column N=lane-16, K = {8..15, 24..31}.
__device__ __forceinline__ v16h load_b_frag(const h16* __restrict__ p, int ld,
                                            int row0, int k0) {
  int lane = threadIdx.x;
  int r  = row0 + (lane & 15);
  int kk = k0 + ((lane & 16) ? 8 : 0);
  const h16* q = p + (size_t)r * ld + kk;
  v8h lo = *(const v8h*)(q);
  v8h hi = *(const v8h*)(q + 16);
  return __builtin_shufflevector(lo, hi, 0,1,2,3,4,5,6,7,8,9,10,11,12,13,14,15);
}

__device__ __forceinline__ v8f wmma_f16(v16h a, v16h b, v8f c) {
  return __builtin_amdgcn_wmma_f32_16x16x32_f16(false, a, false, b,
                                                (short)0, c, false, false);
}

__device__ __forceinline__ float gelu_f(float x) {
  return 0.5f * x * (1.0f + tanhf(0.7978845608028654f * (x + 0.044715f * x * x * x)));
}

// ---------------- generic f16 GEMM, 64x64 tile per wave (4x4 WMMA tiles) ---
// C[M,N] = epi(A[M,K] @ Bt[N,K]^T + bias) + res ; f32 and/or f16 outputs.
// epi: 0=none, 1=gelu, 2=relu, 3=sigmoid.  Stores guarded by m < M.
__device__ __forceinline__ void epilogue_store(v8f acc, const float* __restrict__ bias,
                                               const float* __restrict__ res,
                                               float* __restrict__ Cf,
                                               h16* __restrict__ Ch,
                                               int M, int N, int m0, int n0, int epi) {
  int lane = threadIdx.x;
  int col   = n0 + (lane & 15);
  int mbase = m0 + ((lane & 16) ? 8 : 0);
  float bv = bias ? bias[col] : 0.0f;
#pragma unroll
  for (int r = 0; r < 8; ++r) {
    int m = mbase + r;
    if (m >= M) continue;
    float v = acc[r] + bv;
    if (epi == 1)      v = gelu_f(v);
    else if (epi == 2) v = v > 0.0f ? v : 0.0f;
    else if (epi == 3) v = 1.0f / (1.0f + __expf(-v));
    size_t idx = (size_t)m * N + col;
    if (res) v += res[idx];
    if (Cf) Cf[idx] = v;
    if (Ch) Ch[idx] = (h16)v;
  }
}

__global__ void __launch_bounds__(32)
gemm64x64(const h16* __restrict__ A, const h16* __restrict__ Bt,
          const float* __restrict__ bias, const float* __restrict__ res,
          float* __restrict__ Cf, h16* __restrict__ Ch,
          int M, int N, int K, int epi) {
  int n0 = blockIdx.x * 64;
  int m0 = blockIdx.y * 64;
  v8f acc[4][4];
#pragma unroll
  for (int i = 0; i < 4; ++i)
#pragma unroll
    for (int j = 0; j < 4; ++j) acc[i][j] = (v8f){};

  for (int k0 = 0; k0 < K; k0 += 32) {
    v16h a[4], b[4];
#pragma unroll
    for (int i = 0; i < 4; ++i) a[i] = load_a_frag(A, K, m0 + 16 * i, k0);
#pragma unroll
    for (int j = 0; j < 4; ++j) b[j] = load_b_frag(Bt, K, n0 + 16 * j, k0);
#pragma unroll
    for (int i = 0; i < 4; ++i)
#pragma unroll
      for (int j = 0; j < 4; ++j)
        acc[i][j] = wmma_f16(a[i], b[j], acc[i][j]);
  }
#pragma unroll
  for (int i = 0; i < 4; ++i)
#pragma unroll
    for (int j = 0; j < 4; ++j)
      epilogue_store(acc[i][j], bias, res, Cf, Ch, M, N,
                     m0 + 16 * i, n0 + 16 * j, epi);
}

// ---------------- attention ----------------
// scores[hb,s,s'] = (q/8) . k  ; one 16x16 tile per wave, K = HD = 64
__global__ void __launch_bounds__(32)
attn_scores(const h16* __restrict__ q, const h16* __restrict__ k,
            float* __restrict__ sc) {
  int hb = blockIdx.z;
  int n0 = blockIdx.x * 16, m0 = blockIdx.y * 16;
  const h16* qa = q + (size_t)hb * Sc * HDc;
  const h16* ka = k + (size_t)hb * Sc * HDc;
  v8f acc = {};
  for (int k0 = 0; k0 < HDc; k0 += 32) {
    v16h a = load_a_frag(qa, HDc, m0, k0);
    v16h b = load_b_frag(ka, HDc, n0, k0);
    acc = wmma_f16(a, b, acc);
  }
  int lane = threadIdx.x;
  int col   = n0 + (lane & 15);
  int mbase = m0 + ((lane & 16) ? 8 : 0);
  float* out = sc + (size_t)hb * Sc * Sc;
#pragma unroll
  for (int r = 0; r < 8; ++r) out[(size_t)(mbase + r) * Sc + col] = acc[r];
}

// causal softmax row -> padded f16 probs P[hb, s, 64] (zeros beyond causal/pad)
__global__ void __launch_bounds__(256)
softmax_causal(const float* __restrict__ sc, h16* __restrict__ P) {
  int row = blockIdx.x * blockDim.x + threadIdx.x;
  if (row >= HB * Sc) return;
  int s = row % Sc;
  const float* x = sc + (size_t)row * Sc;
  h16* p = P + (size_t)(row / Sc) * Sc * SPAD + (size_t)s * SPAD;
  float mx = -1e30f;
  for (int j = 0; j <= s; ++j) mx = fmaxf(mx, x[j]);
  float sum = 0.0f;
  for (int j = 0; j <= s; ++j) sum += __expf(x[j] - mx);
  float inv = 1.0f / sum;
  for (int j = 0; j < SPAD; ++j)
    p[j] = (h16)((j <= s) ? __expf(x[j] - mx) * inv : 0.0f);
}

// a[b,s,h*64+d] = P[hb] @ v[hb] ; A=P[48,64pad], Bt=vT[64,64pad]
__global__ void __launch_bounds__(32)
attn_av(const h16* __restrict__ P, const h16* __restrict__ vT,
        h16* __restrict__ aout) {
  int hb = blockIdx.z;
  int n0 = blockIdx.x * 16, m0 = blockIdx.y * 16;   // n over HD, m over S
  const h16* Pa = P  + (size_t)hb * Sc  * SPAD;
  const h16* Va = vT + (size_t)hb * HDc * SPAD;
  v8f acc = {};
  for (int k0 = 0; k0 < SPAD; k0 += 32) {
    v16h a = load_a_frag(Pa, SPAD, m0, k0);
    v16h b = load_b_frag(Va, SPAD, n0, k0);
    acc = wmma_f16(a, b, acc);
  }
  int lane = threadIdx.x;
  int b = hb / NHc, nh = hb % NHc;
  int d     = n0 + (lane & 15);
  int sbase = m0 + ((lane & 16) ? 8 : 0);
#pragma unroll
  for (int r = 0; r < 8; ++r) {
    int s = sbase + r;
    aout[((size_t)(b * Sc + s)) * Hc + nh * HDc + d] = (h16)acc[r];
  }
}

// qkv f32 [B,S,3H] -> q f16 (pre-scaled 1/8), k f16 [hb,S,HD], vT f16 [hb,HD,64pad]
__global__ void __launch_bounds__(256)
pack_qkv(const float* __restrict__ qkv, h16* __restrict__ q,
         h16* __restrict__ k, h16* __restrict__ vT) {
  int idx = blockIdx.x * blockDim.x + threadIdx.x;
  if (idx >= HB * HDc * SPAD) return;
  int s  = idx & 63;
  int d  = (idx >> 6) & 63;
  int hb = idx >> 12;
  int b = hb / NHc, nh = hb % NHc;
  if (s < Sc) {
    const float* base = qkv + ((size_t)(b * Sc + s)) * H3 + nh * HDc + d;
    q[((size_t)hb * Sc + s) * HDc + d]   = (h16)(base[0] * 0.125f);
    k[((size_t)hb * Sc + s) * HDc + d]   = (h16)(base[Hc]);
    vT[((size_t)hb * HDc + d) * SPAD + s] = (h16)(base[2 * Hc]);
  } else {
    vT[((size_t)hb * HDc + d) * SPAD + s] = (h16)0.0f;   // zero K padding
  }
}

// ---------------- group norm over (S, channels-in-group) per (b,g) ----------------
__global__ void __launch_bounds__(256)
group_norm(const float* __restrict__ h, const float* __restrict__ w,
           const float* __restrict__ bb, h16* __restrict__ out) {
  __shared__ float r1[256], r2[256];
  int bid = blockIdx.x;
  int b = bid / Gc, g = bid % Gc;
  int tid = threadIdx.x;
  float s = 0.0f, s2 = 0.0f;
  for (int i = tid; i < Sc * CPG; i += 256) {
    int ss = i / CPG, c = g * CPG + i % CPG;
    float v = h[((size_t)(b * Sc + ss)) * Hc + c];
    s += v; s2 += v * v;
  }
  r1[tid] = s; r2[tid] = s2;
  __syncthreads();
  for (int st = 128; st > 0; st >>= 1) {
    if (tid < st) { r1[tid] += r1[tid + st]; r2[tid] += r2[tid + st]; }
    __syncthreads();
  }
  float mean = r1[0] / (float)(Sc * CPG);
  float var  = r2[0] / (float)(Sc * CPG) - mean * mean;
  float rs = rsqrtf(var + 1e-5f);
  for (int i = tid; i < Sc * CPG; i += 256) {
    int ss = i / CPG, c = g * CPG + i % CPG;
    size_t idx = ((size_t)(b * Sc + ss)) * Hc + c;
    out[idx] = (h16)((h[idx] - mean) * rs * w[c] + bb[c]);
  }
}

// ---------------- elementwise / prep kernels ----------------
// Wt[n, 0..ldk) = f16( W[k, n] ) transposed, zero pad k >= K
__global__ void __launch_bounds__(256)
transpose_to_f16(const float* __restrict__ W, h16* __restrict__ Wt,
                 int K, int N, int ldk, int total) {
  int idx = blockIdx.x * blockDim.x + threadIdx.x;
  if (idx >= total) return;
  int k = idx % ldk, n = idx / ldk;
  Wt[idx] = (h16)((k < K) ? W[(size_t)k * N + n] : 0.0f);
}

// row-major f32 [M,K] -> f16 [M,ldk] with zero K padding
__global__ void __launch_bounds__(256)
convert_pad_f16(const float* __restrict__ X, h16* __restrict__ Y,
                int K, int ldk, int total) {
  int idx = blockIdx.x * blockDim.x + threadIdx.x;
  if (idx >= total) return;
  int k = idx % ldk, m = idx / ldk;
  Y[idx] = (h16)((k < K) ? X[(size_t)m * K + k] : 0.0f);
}

// Wt[n,k] = f16( k<=n ? W[n,k] : 0 )  (tril mask, same layout)
__global__ void __launch_bounds__(256)
convert_tril_f16(const float* __restrict__ W, h16* __restrict__ Wt,
                 int K, int total) {
  int idx = blockIdx.x * blockDim.x + threadIdx.x;
  if (idx >= total) return;
  int k = idx % K, n = idx / K;
  Wt[idx] = (h16)((k <= n) ? W[idx] : 0.0f);
}

__global__ void __launch_bounds__(256)
add_pos(float* __restrict__ h, const float* __restrict__ pos) {
  int idx = blockIdx.x * blockDim.x + threadIdx.x;
  if (idx >= MROWS * Hc) return;
  int c = idx % Hc;
  int s = (idx / Hc) % Sc;
  h[idx] += pos[s * Hc + c];
}

// concat f16: cc[b, t, 0:H]=hn[b,t] ; cc[b, t, H:2H]=hn[b,t+1]
__global__ void __launch_bounds__(256)
build_concat(const h16* __restrict__ hn, h16* __restrict__ cc) {
  int idx = blockIdx.x * blockDim.x + threadIdx.x;
  if (idx >= MH * H2) return;
  int c = idx % H2;
  int row = idx / H2;
  int b = row / (Sc - 1), t = row % (Sc - 1);
  h16 v = (c < Hc) ? hn[((size_t)(b * Sc + t)) * Hc + c]
                   : hn[((size_t)(b * Sc + t + 1)) * Hc + (c - Hc)];
  cc[idx] = v;
}

// ---------------- launcher ----------------
extern "C" void kernel_launch(void* const* d_in, const int* in_sizes, int n_in,
                              void* d_out, int out_size, void* d_ws, size_t ws_size,
                              hipStream_t stream) {
  (void)in_sizes; (void)n_in; (void)out_size; (void)ws_size;
  const float* input_visits = (const float*)d_in[0];
  const float* vis_embed    = (const float*)d_in[1];
  const float* pos_embed    = (const float*)d_in[2];
  const float* ln1_w  = (const float*)d_in[3];
  const float* ln1_b  = (const float*)d_in[4];
  const float* attn_w = (const float*)d_in[5];
  const float* attn_b = (const float*)d_in[6];
  const float* proj_w = (const float*)d_in[7];
  const float* proj_b = (const float*)d_in[8];
  const float* ln2_w  = (const float*)d_in[9];
  const float* ln2_b  = (const float*)d_in[10];
  const float* fc_w   = (const float*)d_in[11];
  const float* fc_b   = (const float*)d_in[12];
  const float* mproj_w = (const float*)d_in[13];
  const float* mproj_b = (const float*)d_in[14];
  const float* lnf_w  = (const float*)d_in[15];
  const float* lnf_b  = (const float*)d_in[16];
  const float* auto1_w = (const float*)d_in[17];
  const float* auto1_b = (const float*)d_in[18];
  const float* auto2_w = (const float*)d_in[19];
  const float* auto2_b = (const float*)d_in[20];

  char* wp = (char*)d_ws;
  auto alloc = [&](size_t bytes) -> char* {
    char* r = wp; wp += (bytes + 255) & ~(size_t)255; return r;
  };
  h16*  Avis   = (h16*)alloc((size_t)MROWS * VPAD * 2);
  h16*  visT   = (h16*)alloc((size_t)Hc * VPAD * 2);
  h16*  attnT  = (h16*)alloc((size_t)NLc * H3 * Hc * 2);
  h16*  projT  = (h16*)alloc((size_t)NLc * Hc * Hc * 2);
  h16*  fcT    = (h16*)alloc((size_t)NLc * H4 * Hc * 2);
  h16*  mprojT = (h16*)alloc((size_t)NLc * Hc * H4 * 2);
  h16*  a1m    = (h16*)alloc((size_t)H2 * H2 * 2);
  h16*  a2m    = (h16*)alloc((size_t)CVc * H2 * 2);
  float* hbuf  = (float*)alloc((size_t)MROWS * Hc * 4);
  h16*  xh     = (h16*)alloc((size_t)MROWS * Hc * 2);
  float* qkvf  = (float*)alloc((size_t)MROWS * H3 * 4);
  h16*  qh     = (h16*)alloc((size_t)HB * Sc * HDc * 2);
  h16*  kh     = (h16*)alloc((size_t)HB * Sc * HDc * 2);
  h16*  vTh    = (h16*)alloc((size_t)HB * HDc * SPAD * 2);
  float* scf   = (float*)alloc((size_t)HB * Sc * Sc * 4);
  h16*  Ph     = (h16*)alloc((size_t)HB * Sc * SPAD * 2);
  h16*  ah     = (h16*)alloc((size_t)MROWS * Hc * 2);
  h16*  midh   = (h16*)alloc((size_t)MROWS * H4 * 2);
  h16*  cch    = (h16*)alloc((size_t)MHPAD * H2 * 2);   // padded rows for M-edge
  h16*  a1h    = (h16*)alloc((size_t)MHPAD * H2 * 2);   // padded rows for M-edge

  const int T = 256;
  dim3 wblk(32);

  // ---- weight prep (f16 convert / transpose / tril mask), every call ----
  convert_pad_f16<<<CEILD(MROWS * VPAD, T), T, 0, stream>>>(
      input_visits, Avis, Vc, VPAD, MROWS * VPAD);
  transpose_to_f16<<<CEILD(Hc * VPAD, T), T, 0, stream>>>(
      vis_embed, visT, Vc, Hc, VPAD, Hc * VPAD);
  for (int i = 0; i < NLc; ++i) {
    transpose_to_f16<<<CEILD(H3 * Hc, T), T, 0, stream>>>(
        attn_w + (size_t)i * Hc * H3, attnT + (size_t)i * H3 * Hc, Hc, H3, Hc, H3 * Hc);
    transpose_to_f16<<<CEILD(Hc * Hc, T), T, 0, stream>>>(
        proj_w + (size_t)i * Hc * Hc, projT + (size_t)i * Hc * Hc, Hc, Hc, Hc, Hc * Hc);
    transpose_to_f16<<<CEILD(H4 * Hc, T), T, 0, stream>>>(
        fc_w + (size_t)i * Hc * H4, fcT + (size_t)i * H4 * Hc, Hc, H4, Hc, H4 * Hc);
    transpose_to_f16<<<CEILD(Hc * H4, T), T, 0, stream>>>(
        mproj_w + (size_t)i * H4 * Hc, mprojT + (size_t)i * Hc * H4, H4, Hc, H4, Hc * H4);
  }
  convert_tril_f16<<<CEILD(H2 * H2, T), T, 0, stream>>>(auto1_w, a1m, H2, H2 * H2);
  convert_tril_f16<<<CEILD(CVc * H2, T), T, 0, stream>>>(auto2_w, a2m, H2, CVc * H2);

  // ---- embedding: h = visits @ vis_embed + pos ----
  gemm64x64<<<dim3(Hc / 64, MROWS / 64), wblk, 0, stream>>>(
      Avis, visT, nullptr, nullptr, hbuf, nullptr, MROWS, Hc, VPAD, 0);
  add_pos<<<CEILD(MROWS * Hc, T), T, 0, stream>>>(hbuf, pos_embed);

  // ---- transformer layers ----
  for (int i = 0; i < NLc; ++i) {
    group_norm<<<Bc * Gc, 256, 0, stream>>>(hbuf, ln1_w + i * Hc, ln1_b + i * Hc, xh);
    gemm64x64<<<dim3(H3 / 64, MROWS / 64), wblk, 0, stream>>>(
        xh, attnT + (size_t)i * H3 * Hc, attn_b + i * H3, nullptr,
        qkvf, nullptr, MROWS, H3, Hc, 0);
    pack_qkv<<<CEILD(HB * HDc * SPAD, T), T, 0, stream>>>(qkvf, qh, kh, vTh);
    attn_scores<<<dim3(Sc / 16, Sc / 16, HB), wblk, 0, stream>>>(qh, kh, scf);
    softmax_causal<<<CEILD(HB * Sc, T), T, 0, stream>>>(scf, Ph);
    attn_av<<<dim3(HDc / 16, Sc / 16, HB), wblk, 0, stream>>>(Ph, vTh, ah);
    gemm64x64<<<dim3(Hc / 64, MROWS / 64), wblk, 0, stream>>>(
        ah, projT + (size_t)i * Hc * Hc, proj_b + i * Hc, hbuf,
        hbuf, nullptr, MROWS, Hc, Hc, 0);
    group_norm<<<Bc * Gc, 256, 0, stream>>>(hbuf, ln2_w + i * Hc, ln2_b + i * Hc, xh);
    gemm64x64<<<dim3(H4 / 64, MROWS / 64), wblk, 0, stream>>>(
        xh, fcT + (size_t)i * H4 * Hc, fc_b + i * H4, nullptr,
        nullptr, midh, MROWS, H4, Hc, 1 /*gelu*/);
    gemm64x64<<<dim3(Hc / 64, MROWS / 64), wblk, 0, stream>>>(
        midh, mprojT + (size_t)i * Hc * H4, mproj_b + i * Hc, hbuf,
        hbuf, nullptr, MROWS, Hc, H4, 0);
  }

  // ---- final norm + autoregressive head ----
  group_norm<<<Bc * Gc, 256, 0, stream>>>(hbuf, lnf_w, lnf_b, xh);
  build_concat<<<CEILD(MH * H2, T), T, 0, stream>>>(xh, cch);
  gemm64x64<<<dim3(H2 / 64, CEILD(MH, 64)), wblk, 0, stream>>>(
      cch, a1m, auto1_b, nullptr, nullptr, a1h, MH, H2, H2, 2 /*relu*/);
  gemm64x64<<<dim3(CVc / 64, CEILD(MH, 64)), wblk, 0, stream>>>(
      a1h, a2m, auto2_b, nullptr, (float*)d_out, nullptr, MH, CVc, H2, 3 /*sigmoid*/);
}
